// ConvRNN_33354716021156
// MI455X (gfx1250) — matrix operand order repaired and checked by reading, not compiled, not run on hardware
//
#include <hip/hip_runtime.h>
#include <hip/hip_bf16.h>

typedef __bf16  v16bf __attribute__((ext_vector_type(16)));
typedef float   v8f   __attribute__((ext_vector_type(8)));
typedef float   v4f   __attribute__((ext_vector_type(4)));

// ---------------------------------------------------------------------------
// Elementwise / graph-normalization kernels
// ---------------------------------------------------------------------------
__global__ void deg_init_kernel(float* __restrict__ deg, int n) {
  int i = blockIdx.x * blockDim.x + threadIdx.x;
  if (i < n) deg[i] = 1.0f;                      // self-loop contributes 1
}

__global__ void deg_accum_kernel(const long long* __restrict__ dst,
                                 float* __restrict__ deg, int e) {
  int i = blockIdx.x * blockDim.x + threadIdx.x;
  if (i < e) unsafeAtomicAdd(&deg[dst[i]], 1.0f);
}

__global__ void dinv_kernel(float* __restrict__ deg, int n) {
  int i = blockIdx.x * blockDim.x + threadIdx.x;
  if (i < n) deg[i] = __frsqrt_rn(deg[i]);       // deg >= 1 always
}

// ---------------------------------------------------------------------------
// Pack a (KT*32 x 128) row-major f32 weight matrix into bf16 B-fragments.
// Layout per (ntile, ktile): 32 lanes x 16 bf16, where for lane L, element i
// holds W[kt*32 + 8*(L>>4) + (i&7) + 16*(i>>3)][nt*16 + (L&15)]  (ISA 7.12.2 B layout).
// ---------------------------------------------------------------------------
__global__ void pack_b_kernel(const float* __restrict__ W, __bf16* __restrict__ out,
                              int KT) {
  int idx = blockIdx.x * blockDim.x + threadIdx.x;
  int total = 8 * KT * 32 * 16;
  if (idx >= total) return;
  int i    =  idx       & 15;
  int lane = (idx >> 4) & 31;
  int t    =  idx >> 9;              // nt*KT + kt
  int kt   =  t % KT;
  int nt   =  t / KT;
  int k = kt * 32 + ((lane >> 4) << 3) + (i & 7) + ((i >> 3) << 4);
  int n = nt * 16 + (lane & 15);
  out[idx] = (__bf16)W[(size_t)k * 128 + n];
}

// ---------------------------------------------------------------------------
// Build a 16x32 bf16 A-fragment from a row-major f32 row pointer.
// Lane L (row M = L&15) needs K = kb..kb+7 and kb+16..kb+23, kb = 8*(L>>4)
// (plus the caller's 32*kt k-tile base folded into `kb`).
// ---------------------------------------------------------------------------
__device__ __forceinline__ v16bf load_a_frag_f32(const float* __restrict__ rowp, int kb) {
  v4f a0 = *(const v4f*)(rowp + kb);
  v4f a1 = *(const v4f*)(rowp + kb + 4);
  v4f a2 = *(const v4f*)(rowp + kb + 16);
  v4f a3 = *(const v4f*)(rowp + kb + 20);
  v16bf r;
  r[0]  = (__bf16)a0[0]; r[1]  = (__bf16)a0[1]; r[2]  = (__bf16)a0[2]; r[3]  = (__bf16)a0[3];
  r[4]  = (__bf16)a1[0]; r[5]  = (__bf16)a1[1]; r[6]  = (__bf16)a1[2]; r[7]  = (__bf16)a1[3];
  r[8]  = (__bf16)a2[0]; r[9]  = (__bf16)a2[1]; r[10] = (__bf16)a2[2]; r[11] = (__bf16)a2[3];
  r[12] = (__bf16)a3[0]; r[13] = (__bf16)a3[1]; r[14] = (__bf16)a3[2]; r[15] = (__bf16)a3[3];
  return r;
}

// ---------------------------------------------------------------------------
// GEMM1: h[N,128] = concat(x, hidden)[N,256] @ W[256,128]
// One wave computes a full 16-row x 128-col strip (8 f32 16x16 accumulators).
// Per k-tile: stage all 8 B fragments (one 16-load clause), prefetch the next
// A fragment, then issue the 8-WMMA burst with staggered load waits.
// ---------------------------------------------------------------------------
__global__ __launch_bounds__(256)
void gemm1_kernel(const float* __restrict__ x, const float* __restrict__ hs,
                  const __bf16* __restrict__ Wp, float* __restrict__ hout, int nTiles) {
  int wave = (int)((blockIdx.x * blockDim.x + threadIdx.x) >> 5);
  if (wave >= nTiles) return;                      // whole-wave exit: EXEC stays all-1
  int lane = threadIdx.x & 31;
  int hi   = lane >> 4;
  int m0   = wave << 4;
  int row  = m0 + (lane & 15);
  const float* xrow = x  + (size_t)row * 128;
  const float* hrow = hs + (size_t)row * 128;
  v8f acc[8] = {};
  v16bf a = load_a_frag_f32(xrow, hi * 8);         // kt = 0 prologue
#pragma unroll
  for (int kt = 0; kt < 8; ++kt) {
    v16bf bfrag[8];
#pragma unroll
    for (int nt = 0; nt < 8; ++nt)                 // one big load clause
      bfrag[nt] = *(const v16bf*)(Wp + (((size_t)nt * 8 + kt) * 32 + lane) * 16);
    v16bf anext;
    if (kt < 7) {                                  // prefetch next A fragment
      const float* ap = (kt + 1 < 4) ? xrow : hrow;
      anext = load_a_frag_f32(ap, ((kt + 1) & 3) * 32 + hi * 8);
    }
#pragma unroll
    for (int nt = 0; nt < 8; ++nt)
      acc[nt] = __builtin_amdgcn_wmma_f32_16x16x32_bf16(
          false, a, false, bfrag[nt], (short)0, acc[nt], false, false);
    if (kt < 7) a = anext;
  }
  int col0 = lane & 15;
#pragma unroll
  for (int nt = 0; nt < 8; ++nt)
#pragma unroll
    for (int r = 0; r < 8; ++r)
      hout[(size_t)(m0 + r + 8 * hi) * 128 + nt * 16 + col0] = acc[nt][r];
}

// ---------------------------------------------------------------------------
// agg = h * dinv^2  (self-loop term; also fully re-initializes agg each call)
// ---------------------------------------------------------------------------
__global__ void agg_init_kernel(const float* __restrict__ h, const float* __restrict__ dinv,
                                float* __restrict__ agg, long long total) {
  long long i = (long long)blockIdx.x * blockDim.x + threadIdx.x;
  if (i < total) { float di = dinv[i >> 7]; agg[i] = h[i] * di * di; }
}

// ---------------------------------------------------------------------------
// Edge scatter: one wave per edge; src/dst are wave-uniform (scalar loads);
// each lane gathers 4 contiguous floats of h[src], scales, and does 4
// global_atomic_add_f32 into agg[dst] (resolved in the 192MB L2).
// ---------------------------------------------------------------------------
__global__ __launch_bounds__(256)
void scatter_kernel(const long long* __restrict__ src, const long long* __restrict__ dst,
                    const float* __restrict__ dinv, const float* __restrict__ h,
                    float* __restrict__ agg, int e) {
  int eid = blockIdx.x * 8 + (threadIdx.x >> 5);
  if (eid >= e) return;
  int lane = threadIdx.x & 31;
  long long s = src[eid];
  long long d = dst[eid];
  float norm = dinv[s] * dinv[d];
  v4f v = *((const v4f*)(h + ((size_t)s << 7)) + lane);
  float* ap = agg + ((size_t)d << 7) + lane * 4;
  unsafeAtomicAdd(ap + 0, v[0] * norm);
  unsafeAtomicAdd(ap + 1, v[1] * norm);
  unsafeAtomicAdd(ap + 2, v[2] * norm);
  unsafeAtomicAdd(ap + 3, v[3] * norm);
}

// ---------------------------------------------------------------------------
// new_hidden = sigmoid(b_matrix + agg + bias)
// ---------------------------------------------------------------------------
__global__ void sigmoid_kernel(const float* __restrict__ agg, const float* __restrict__ bias,
                               const float* __restrict__ bmat, float* __restrict__ nh,
                               long long total) {
  long long i = (long long)blockIdx.x * blockDim.x + threadIdx.x;
  if (i < total) {
    float a = agg[i] + bias[i & 127] + bmat[i];
    nh[i] = 1.0f / (1.0f + __expf(-a));
  }
}

// ---------------------------------------------------------------------------
// GEMM2: o[N,128] = new_hidden[N,128] @ V[128,128] + c_matrix
// ---------------------------------------------------------------------------
__global__ __launch_bounds__(256)
void gemm2_kernel(const float* __restrict__ nh, const __bf16* __restrict__ Vp,
                  const float* __restrict__ cmat, float* __restrict__ o, int nTiles) {
  int wave = (int)((blockIdx.x * blockDim.x + threadIdx.x) >> 5);
  if (wave >= nTiles) return;
  int lane = threadIdx.x & 31;
  int hi   = lane >> 4;
  int m0   = wave << 4;
  int row  = m0 + (lane & 15);
  const float* arow = nh + (size_t)row * 128;
  v8f acc[8] = {};
  v16bf a = load_a_frag_f32(arow, hi * 8);         // kt = 0 prologue
#pragma unroll
  for (int kt = 0; kt < 4; ++kt) {
    v16bf bfrag[8];
#pragma unroll
    for (int nt = 0; nt < 8; ++nt)
      bfrag[nt] = *(const v16bf*)(Vp + (((size_t)nt * 4 + kt) * 32 + lane) * 16);
    v16bf anext;
    if (kt < 3) anext = load_a_frag_f32(arow, (kt + 1) * 32 + hi * 8);
#pragma unroll
    for (int nt = 0; nt < 8; ++nt)
      acc[nt] = __builtin_amdgcn_wmma_f32_16x16x32_bf16(
          false, a, false, bfrag[nt], (short)0, acc[nt], false, false);
    if (kt < 3) a = anext;
  }
  int col0 = lane & 15;
#pragma unroll
  for (int nt = 0; nt < 8; ++nt)
#pragma unroll
    for (int r = 0; r < 8; ++r) {
      size_t idx = (size_t)(m0 + r + 8 * hi) * 128 + nt * 16 + col0;
      o[idx] = acc[nt][r] + cmat[idx];
    }
}

// ---------------------------------------------------------------------------
// Host-side launch
// ---------------------------------------------------------------------------
extern "C" void kernel_launch(void* const* d_in, const int* in_sizes, int n_in,
                              void* d_out, int out_size, void* d_ws, size_t ws_size,
                              hipStream_t stream) {
  const float*     x    = (const float*)d_in[0];
  const float*     hs   = (const float*)d_in[1];
  const float*     W    = (const float*)d_in[2];   // [256,128]
  const float*     bias = (const float*)d_in[3];   // [128]
  const float*     bmat = (const float*)d_in[4];   // [N,128]
  const float*     V    = (const float*)d_in[5];   // [128,128]
  const float*     cmat = (const float*)d_in[6];   // [N,128]
  const long long* ei   = (const long long*)d_in[7]; // [2,E] int64

  const int N = in_sizes[0] / 128;
  const int E = in_sizes[7] / 2;
  const long long* srcp = ei;
  const long long* dstp = ei + E;

  // workspace carve-out (256B aligned)
  char* ws = (char*)d_ws;
  size_t off = 0;
  auto take = [&](size_t bytes) {
    size_t o = off;
    off += (bytes + 255) & ~(size_t)255;
    return o;
  };
  __bf16* Wp   = (__bf16*)(ws + take((size_t)8 * 8 * 32 * 16 * 2)); // 64 KB
  __bf16* Vp   = (__bf16*)(ws + take((size_t)8 * 4 * 32 * 16 * 2)); //  32 KB
  float*  deg  = (float*) (ws + take((size_t)N * 4));               // deg -> dinv in place
  float*  hbuf = (float*) (ws + take((size_t)N * 128 * 4));         // 51.2 MB
  float*  agg  = (float*) (ws + take((size_t)N * 128 * 4));         // 51.2 MB

  float* o_out = (float*)d_out;                  // o  : first N*128 floats
  float* nh    = o_out + (size_t)N * 128;        // new_hidden : second N*128 floats

  const long long total = (long long)N * 128;
  const int mt = N / 16;                          // 100000 / 16 = 6250 exact

  deg_init_kernel <<<(N + 255) / 256, 256, 0, stream>>>(deg, N);
  deg_accum_kernel<<<(E + 255) / 256, 256, 0, stream>>>(dstp, deg, E);
  dinv_kernel     <<<(N + 255) / 256, 256, 0, stream>>>(deg, N);

  pack_b_kernel<<<(8 * 8 * 32 * 16 + 255) / 256, 256, 0, stream>>>(W, Wp, 8);
  pack_b_kernel<<<(8 * 4 * 32 * 16 + 255) / 256, 256, 0, stream>>>(V, Vp, 4);

  gemm1_kernel<<<(mt * 32 + 255) / 256, 256, 0, stream>>>(x, hs, Wp, hbuf, mt);

  agg_init_kernel<<<(unsigned)((total + 255) / 256), 256, 0, stream>>>(hbuf, deg, agg, total);
  scatter_kernel <<<(E + 7) / 8, 256, 0, stream>>>(srcp, dstp, deg, hbuf, agg, E);

  sigmoid_kernel<<<(unsigned)((total + 255) / 256), 256, 0, stream>>>(agg, bias, bmat, nh, total);

  gemm2_kernel<<<(mt * 32 + 255) / 256, 256, 0, stream>>>(nh, Vp, cmat, o_out, mt);
}
